// ConceptModelGPT2_90787018703483
// MI455X (gfx1250) — matrix operand order repaired and checked by reading, not compiled, not run on hardware
//
#include <hip/hip_runtime.h>
#include <math.h>

// ---------------- model dims (hardcoded from reference) ----------------
constexpr int Bd  = 8;
constexpr int Td  = 1024;
constexpr int DAE = 512;
constexpr int Dm  = 768;
constexpr int Ld  = 12;
constexpr int Hh  = 12;
constexpr int HDd = 64;
constexpr int QKVS = 3 * Dm;          // 2304
constexpr int MTOK = Bd * Td;         // 8192 rows

typedef unsigned short u16;
typedef unsigned int   u32;
typedef unsigned long long u64;
typedef __attribute__((ext_vector_type(16))) __bf16 v16bf;
typedef __attribute__((ext_vector_type(8)))  float  v8f;

union Frag16 { v16bf v; u16 u[16]; u32 d[8]; };

__device__ __forceinline__ u16 f2bf(float f) {
  u32 u = __float_as_uint(f);
  u += 0x7FFFu + ((u >> 16) & 1u);      // round-to-nearest-even
  return (u16)(u >> 16);
}

// ---------------- fp32 -> bf16 transposing weight conversion ----------------
// src: K x N row-major f32  ->  dst: N x K row-major bf16 (i.e. B^T)
__global__ __launch_bounds__(256)
void cvtT_kernel(const float* __restrict__ src, u16* __restrict__ dst, int K, int N) {
  __shared__ float t[32][33];
  const int kb = blockIdx.x * 32, nb = blockIdx.y * 32;
  const int tx = threadIdx.x & 31, ty = threadIdx.x >> 5;   // ty: 0..7
  #pragma unroll
  for (int i = ty; i < 32; i += 8) t[i][tx] = src[(size_t)(kb + i) * N + nb + tx];
  __syncthreads();
  #pragma unroll
  for (int i = ty; i < 32; i += 8) dst[(size_t)(nb + i) * K + kb + tx] = f2bf(t[tx][i]);
}

// ---------------- LayerNorm (row-wise), bf16 or f32 output ----------------
__global__ __launch_bounds__(256)
void ln_kernel(const float* __restrict__ X, const float* __restrict__ g,
               const float* __restrict__ b, void* __restrict__ out,
               int C, int out_is_f32) {
  const int row = blockIdx.x;
  const int tid = threadIdx.x;
  const float* x = X + (size_t)row * C;
  float s = 0.f, s2 = 0.f;
  for (int k = tid; k < C; k += 256) { float v = x[k]; s += v; s2 += v * v; }
  for (int off = 16; off > 0; off >>= 1) {
    s  += __shfl_down(s,  off);
    s2 += __shfl_down(s2, off);
  }
  __shared__ float red0[8], red1[8];
  const int wave = tid >> 5, lane = tid & 31;
  if (lane == 0) { red0[wave] = s; red1[wave] = s2; }
  __syncthreads();
  float ts = 0.f, ts2 = 0.f;
  #pragma unroll
  for (int i = 0; i < 8; i++) { ts += red0[i]; ts2 += red1[i]; }
  const float mean = ts / C;
  const float var  = ts2 / C - mean * mean;
  const float inv  = rsqrtf(var + 1e-5f);
  if (out_is_f32) {
    float* o = (float*)out + (size_t)row * C;
    for (int k = tid; k < C; k += 256)
      o[k] = (x[k] - mean) * inv * g[k] + b[k];
  } else {
    u16* o = (u16*)out + (size_t)row * C;
    for (int k = tid; k < C; k += 256)
      o[k] = f2bf((x[k] - mean) * inv * g[k] + b[k]);
  }
}

// ---------------- generic bf16 WMMA GEMM ----------------
// C[M,N] = A[M,K](bf16,row-major) * B^T[N,K](bf16,row-major) + epilogue
// A staged via async-to-LDS double buffering (BK=64 deep stages); B fragments
// loaded directly from global (weights are L2-resident: 171 MB < 192 MB L2).
enum { EP_F32 = 0, EP_F32_RES = 1, EP_BF16_GELU = 2, EP_BF16 = 3, EP_F32_POS = 4 };

#define BM 128
#define BN 128
#define BK 64

__global__ __launch_bounds__(256)
void gemm_bf16_kernel(const u16* __restrict__ A, const u16* __restrict__ BwT,
                      const float* __restrict__ bias,
                      const float* __restrict__ resp,   // residual or wpe
                      void* __restrict__ Cout,
                      int M, int N, int K, int ep) {
  __shared__ __align__(16) u16 lsA[2][BM * BK];   // double-buffered 16 KB A tiles
  const int tid  = threadIdx.x;
  const int wave = tid >> 5, lane = tid & 31;
  const int g = lane >> 4, lr = lane & 15;
  const int nTilesN = N / BN;
  const int m0 = (blockIdx.x / nTilesN) * BM;
  const int n0 = (blockIdx.x % nTilesN) * BN;
  const int wm = wave & 3;    // 4 strips of 32 rows
  const int wn = wave >> 2;   // 2 strips of 64 cols

  const v8f vzero = {0.f, 0.f, 0.f, 0.f, 0.f, 0.f, 0.f, 0.f};
  v8f acc[2][4];
  #pragma unroll
  for (int i = 0; i < 2; i++)
    #pragma unroll
    for (int j = 0; j < 4; j++) acc[i][j] = vzero;

  // Async stage one 128x64 A tile: 16 KB; each thread issues 4 x b128 (16 B).
  auto issueA = [&](int k0, int buf) {
    #pragma unroll
    for (int q = 0; q < 4; q++) {
      const int c  = tid + q * 256;          // 1024 chunks of 16 B
      const int m  = c >> 3;                 // 8 chunks per 64-elem row
      const int kc = (c & 7) * 8;            // bf16 index within row
      const u16* gp = A + (size_t)(m0 + m) * K + k0 + kc;
      const u32 laddr = (u32)(uintptr_t)&lsA[buf][m * BK + kc];
      asm volatile("global_load_async_to_lds_b128 %0, %1, off"
                   :: "v"(laddr), "v"((u64)(uintptr_t)gp) : "memory");
    }
  };

  issueA(0, 0);
  int cur = 0;
  for (int k0 = 0; k0 < K; k0 += BK) {
    const bool hasNext = (k0 + BK) < K;
    if (hasNext) {
      issueA(k0 + BK, cur ^ 1);
      asm volatile("s_wait_asynccnt 0x4" ::: "memory");  // next tile stays in flight
    } else {
      asm volatile("s_wait_asynccnt 0x0" ::: "memory");
    }
    __syncthreads();

    // Preload ALL B fragments for both K=32 substeps (independent registers,
    // lets the scheduler clause the 16 b128 loads and cover them with WMMAs).
    Frag16 bfr[2][4];
    #pragma unroll
    for (int ks = 0; ks < 2; ks++) {
      #pragma unroll
      for (int nt = 0; nt < 4; nt++) {
        const u32* pb = (const u32*)(BwT + (size_t)(n0 + wn * 64 + nt * 16 + lr) * K
                                     + k0 + ks * 32);
        #pragma unroll
        for (int i = 0; i < 8; i++) bfr[ks][nt].d[i] = pb[8 * g + i];
        if (ks == 0) __builtin_prefetch((const void*)(pb + 32), 0, 3);  // next stage
      }
    }
    #pragma unroll
    for (int ks = 0; ks < 2; ks++) {        // two K=32 substeps per staged tile
      // A fragments from LDS
      Frag16 afr[2];
      #pragma unroll
      for (int mt = 0; mt < 2; mt++) {
        const u32* pa = (const u32*)&lsA[cur][(wm * 32 + mt * 16 + lr) * BK + ks * 32];
        #pragma unroll
        for (int i = 0; i < 8; i++) afr[mt].d[i] = pa[g * 8 + i];
      }
      #pragma unroll
      for (int mt = 0; mt < 2; mt++)
        #pragma unroll
        for (int nt = 0; nt < 4; nt++)
          acc[mt][nt] = __builtin_amdgcn_wmma_f32_16x16x32_bf16(
              false, afr[mt].v, false, bfr[ks][nt].v, (short)0, acc[mt][nt], false, false);
    }
    __syncthreads();
    cur ^= 1;
  }

  // epilogue
  #pragma unroll
  for (int mt = 0; mt < 2; mt++) {
    #pragma unroll
    for (int nt = 0; nt < 4; nt++) {
      const int c = n0 + wn * 64 + nt * 16 + lr;
      const float bv = bias ? bias[c] : 0.f;
      #pragma unroll
      for (int r = 0; r < 8; r++) {
        const int row = m0 + wm * 32 + mt * 16 + 8 * g + r;
        float v = acc[mt][nt][r] + bv;
        const size_t idx = (size_t)row * N + c;
        if (ep == EP_F32) {
          ((float*)Cout)[idx] = v;
        } else if (ep == EP_F32_RES) {
          ((float*)Cout)[idx] = v + resp[idx];
        } else if (ep == EP_F32_POS) {
          ((float*)Cout)[idx] = v + resp[(size_t)(row & (Td - 1)) * N + c];
        } else if (ep == EP_BF16_GELU) {
          const float t = 0.7978845608028654f * (v + 0.044715f * v * v * v);
          v = 0.5f * v * (1.f + tanhf(t));
          ((u16*)Cout)[idx] = f2bf(v);
        } else { // EP_BF16
          ((u16*)Cout)[idx] = f2bf(v);
        }
      }
    }
  }
}

// ---------------- block-causal flash attention (WMMA) ----------------
// qkv: (B*T, 3*D) bf16 rows = [q | k | v], head-major h*64+d inside each.
// out: (B*T, D) bf16.  One wave handles one 16-row query tile of one (b,h).
__global__ __launch_bounds__(256)
void attn_kernel(const u16* __restrict__ qkv, u16* __restrict__ out) {
  const int qtiles = Td / 128;
  const int qblk = blockIdx.x % qtiles;
  const int bh   = blockIdx.x / qtiles;
  const int h = bh % Hh, b = bh / Hh;
  const int wave = threadIdx.x >> 5, lane = threadIdx.x & 31;
  const int g = lane >> 4, lr = lane & 15;
  const int q0 = qblk * 128 + wave * 16;

  __shared__ __align__(16) u16 lsP[8][16 * 32];   // per-wave P tile (16x32)

  const u16* base = qkv + (size_t)b * Td * QKVS + h * HDd;
  const float scale = 0.125f;  // 1/sqrt(64)

  // Q a-fragments for d=0..31 and d=32..63
  Frag16 aq[2];
  {
    const u32* qp = (const u32*)(base + (size_t)(q0 + lr) * QKVS);
    #pragma unroll
    for (int f = 0; f < 2; f++) {
      const int bo = f * 16 + g * 8;
      #pragma unroll
      for (int i = 0; i < 8; i++) aq[f].d[i] = qp[bo + i];
    }
  }

  const v8f vzero = {0.f, 0.f, 0.f, 0.f, 0.f, 0.f, 0.f, 0.f};
  v8f o[4] = {vzero, vzero, vzero, vzero};
  float mrow[8], lrow[8];
  #pragma unroll
  for (int r = 0; r < 8; r++) { mrow[r] = -1e30f; lrow[r] = 0.f; }

  const int nk = q0 + 16;  // exclusive key limit for this tile (n_latent=8)

  for (int j0 = 0; j0 < nk; j0 += 32) {
    // ---- S = Q * K^T for 32 keys (two 16-wide tiles) ----
    v8f sAcc[2] = {vzero, vzero};
    #pragma unroll
    for (int kt = 0; kt < 2; kt++) {
      int j = j0 + kt * 16 + lr;
      if (j >= nk) j = nk - 1;                       // masked later
      const u32* kp = (const u32*)(base + Dm + (size_t)j * QKVS);
      #pragma unroll
      for (int f = 0; f < 2; f++) {
        Frag16 bk;
        #pragma unroll
        for (int i = 0; i < 8; i++) bk.d[i] = kp[f * 16 + 8 * g + i];
        sAcc[kt] = __builtin_amdgcn_wmma_f32_16x16x32_bf16(
            false, aq[f].v, false, bk.v, (short)0, sAcc[kt], false, false);
      }
    }
    // ---- online softmax over these 32 columns ----
    #pragma unroll
    for (int r = 0; r < 8; r++) {
      const int row = q0 + 8 * g + r;
      const int lim = ((row >> 3) + 1) << 3;         // block-causal limit
      float v0 = (j0 + lr      < lim) ? sAcc[0][r] * scale : -1e30f;
      float v1 = (j0 + 16 + lr < lim) ? sAcc[1][r] * scale : -1e30f;
      float mx = fmaxf(v0, v1);
      mx = fmaxf(mx, __shfl_xor(mx, 1));
      mx = fmaxf(mx, __shfl_xor(mx, 2));
      mx = fmaxf(mx, __shfl_xor(mx, 4));
      mx = fmaxf(mx, __shfl_xor(mx, 8));
      const float mnew  = fmaxf(mrow[r], mx);
      const float alpha = __expf(mrow[r] - mnew);
      mrow[r] = mnew;
      const float p0 = __expf(v0 - mnew);
      const float p1 = __expf(v1 - mnew);
      float ps = p0 + p1;
      ps += __shfl_xor(ps, 1);
      ps += __shfl_xor(ps, 2);
      ps += __shfl_xor(ps, 4);
      ps += __shfl_xor(ps, 8);
      lrow[r] = lrow[r] * alpha + ps;
      #pragma unroll
      for (int c = 0; c < 4; c++) o[c][r] *= alpha;
      lsP[wave][(8 * g + r) * 32 + lr]      = f2bf(p0);
      lsP[wave][(8 * g + r) * 32 + 16 + lr] = f2bf(p1);
    }
    asm volatile("s_wait_dscnt 0x0" ::: "memory");   // wave-private LDS RAW

    // ---- O += P * V ----
    Frag16 ap;
    {
      const u32* pp = (const u32*)&lsP[wave][lr * 32];
      #pragma unroll
      for (int i = 0; i < 8; i++) ap.d[i] = pp[g * 8 + i];
    }
    #pragma unroll
    for (int nt = 0; nt < 4; nt++) {
      Frag16 bv;
      const int d = nt * 16 + lr;
      #pragma unroll
      for (int i = 0; i < 16; i++) {
        int j = j0 + 16 * g + i;
        if (j >= nk) j = nk - 1;                     // p==0 there anyway
        bv.u[i] = base[(size_t)j * QKVS + 2 * Dm + d];
      }
      o[nt] = __builtin_amdgcn_wmma_f32_16x16x32_bf16(
          false, ap.v, false, bv.v, (short)0, o[nt], false, false);
    }
  }

  // ---- finalize: divide by l, store bf16 ----
  #pragma unroll
  for (int r = 0; r < 8; r++) {
    const float inv = 1.0f / lrow[r];
    const size_t row = (size_t)(b * Td + q0 + 8 * g + r);
    #pragma unroll
    for (int nt = 0; nt < 4; nt++)
      out[row * Dm + h * HDd + nt * 16 + lr] = f2bf(o[nt][r] * inv);
  }
}

// ---------------- host: orchestrate the full forward ----------------
extern "C" void kernel_launch(void* const* d_in, const int* in_sizes, int n_in,
                              void* d_out, int out_size, void* d_ws, size_t ws_size,
                              hipStream_t stream) {
  (void)in_sizes; (void)n_in; (void)out_size; (void)ws_size;
  const float* x     = (const float*)d_in[0];
  const float* in_g  = (const float*)d_in[1];
  const float* in_b  = (const float*)d_in[2];
  const float* W_in  = (const float*)d_in[3];
  const float* wpe   = (const float*)d_in[4];
  const float* ln1_g = (const float*)d_in[5];
  const float* ln1_b = (const float*)d_in[6];
  const float* W_qkv = (const float*)d_in[7];
  const float* b_qkv = (const float*)d_in[8];
  const float* W_o   = (const float*)d_in[9];
  const float* b_o   = (const float*)d_in[10];
  const float* ln2_g = (const float*)d_in[11];
  const float* ln2_b = (const float*)d_in[12];
  const float* W_fc  = (const float*)d_in[13];
  const float* b_fc  = (const float*)d_in[14];
  const float* W_pr  = (const float*)d_in[15];
  const float* b_pr  = (const float*)d_in[16];
  const float* lnf_g = (const float*)d_in[17];
  const float* lnf_b = (const float*)d_in[18];
  const float* W_out = (const float*)d_in[19];
  const float* out_g = (const float*)d_in[20];
  const float* out_b = (const float*)d_in[21];
  // d_in[22] = n_latent (== 8, hardcoded)

  // workspace carve (~326 MB total)
  char* p = (char*)d_ws;
  auto carve = [&](size_t bytes) -> void* {
    void* r = (void*)p;
    p += (bytes + 255) & ~(size_t)255;
    return r;
  };
  // all weights stored TRANSPOSED (N x K) in bf16
  u16* wb_in  = (u16*)carve((size_t)DAE * Dm * 2);
  u16* wb_qkv = (u16*)carve((size_t)Ld * Dm * QKVS * 2);
  u16* wb_o   = (u16*)carve((size_t)Ld * Dm * Dm * 2);
  u16* wb_fc  = (u16*)carve((size_t)Ld * Dm * 4 * Dm * 2);
  u16* wb_pr  = (u16*)carve((size_t)Ld * 4 * Dm * Dm * 2);
  u16* wb_out = (u16*)carve((size_t)Dm * DAE * 2);
  float* hbuf = (float*)carve((size_t)MTOK * Dm * 4);
  u16* abf    = (u16*)carve((size_t)MTOK * Dm * 2);
  u16* qkvbf  = (u16*)carve((size_t)MTOK * QKVS * 2);
  u16* attbf  = (u16*)carve((size_t)MTOK * Dm * 2);
  u16* mbf    = (u16*)carve((size_t)MTOK * 4 * Dm * 2);
  float* tmpf = (float*)carve((size_t)MTOK * DAE * 4);

  auto cvtT = [&](const float* src, u16* dst, int K, int N) {
    cvtT_kernel<<<dim3(K / 32, N / 32), dim3(256), 0, stream>>>(src, dst, K, N);
  };
  auto ln = [&](const float* X, const float* gg, const float* bb, void* o, int C, int f32o) {
    ln_kernel<<<dim3(MTOK), dim3(256), 0, stream>>>(X, gg, bb, o, C, f32o);
  };
  auto gemm = [&](const u16* A, const u16* BwT, const float* bias, const float* resp,
                  void* Cout, int M, int N, int K, int ep) {
    gemm_bf16_kernel<<<dim3((M / BM) * (N / BN)), dim3(256), 0, stream>>>(
        A, BwT, bias, resp, Cout, M, N, K, ep);
  };

  // --- weight conversion + transpose (bf16 B^T stays L2-resident: 171 MB) ---
  cvtT(W_in, wb_in, DAE, Dm);
  for (int l = 0; l < Ld; l++) {
    cvtT(W_qkv + (size_t)l * Dm * QKVS,    wb_qkv + (size_t)l * Dm * QKVS,    Dm, QKVS);
    cvtT(W_o   + (size_t)l * Dm * Dm,      wb_o   + (size_t)l * Dm * Dm,      Dm, Dm);
    cvtT(W_fc  + (size_t)l * Dm * 4 * Dm,  wb_fc  + (size_t)l * Dm * 4 * Dm,  Dm, 4 * Dm);
    cvtT(W_pr  + (size_t)l * 4 * Dm * Dm,  wb_pr  + (size_t)l * 4 * Dm * Dm,  4 * Dm, Dm);
  }
  cvtT(W_out, wb_out, Dm, DAE);

  // --- h = LN(x) @ W_in + wpe ---
  ln(x, in_g, in_b, abf, DAE, 0);
  gemm(abf, wb_in, nullptr, wpe, hbuf, MTOK, Dm, DAE, EP_F32_POS);

  // --- transformer layers ---
  for (int l = 0; l < Ld; l++) {
    ln(hbuf, ln1_g + l * Dm, ln1_b + l * Dm, abf, Dm, 0);
    gemm(abf, wb_qkv + (size_t)l * Dm * QKVS, b_qkv + l * QKVS, nullptr,
         qkvbf, MTOK, QKVS, Dm, EP_BF16);
    attn_kernel<<<dim3(Bd * Hh * (Td / 128)), dim3(256), 0, stream>>>(qkvbf, attbf);
    gemm(attbf, wb_o + (size_t)l * Dm * Dm, b_o + l * Dm, hbuf,
         hbuf, MTOK, Dm, Dm, EP_F32_RES);
    ln(hbuf, ln2_g + l * Dm, ln2_b + l * Dm, abf, Dm, 0);
    gemm(abf, wb_fc + (size_t)l * Dm * 4 * Dm, b_fc + l * 4 * Dm, nullptr,
         mbf, MTOK, 4 * Dm, Dm, EP_BF16_GELU);
    gemm(mbf, wb_pr + (size_t)l * 4 * Dm * Dm, b_pr + l * Dm, hbuf,
         hbuf, MTOK, Dm, 4 * Dm, EP_F32_RES);
  }

  // --- out = LN(LN(h) @ W_out) ---
  ln(hbuf, lnf_g, lnf_b, abf, Dm, 0);
  gemm(abf, wb_out, nullptr, nullptr, tmpf, MTOK, DAE, Dm, EP_F32);
  ln(tmpf, out_g, out_b, d_out, DAE, 1);
}
